// AttentionConv3D_VGG_31464930410848
// MI455X (gfx1250) — matrix operand order repaired and checked by reading, not compiled, not run on hardware
//
#include <hip/hip_runtime.h>
#include <hip/hip_bf16.h>

// ---------------------------------------------------------------------------
// CDNA5 (gfx1250) implementation of dual-branch channel attention.
// All GEMM-like work (1x1 convs, channel gram) runs on v_wmma_f32_16x16x32_f16
// with f32 accumulation; depthwise convs / softmax / apply are bandwidth-bound
// VALU kernels. wave32 everywhere.
// ---------------------------------------------------------------------------

typedef __attribute__((ext_vector_type(16))) _Float16 v16h;
typedef __attribute__((ext_vector_type(8)))  float    v8f;

union FragU { v16h v; uint4 u[2]; };

__device__ __forceinline__ v8f v8f_zero() {
  v8f z;
#pragma unroll
  for (int i = 0; i < 8; ++i) z[i] = 0.0f;
  return z;
}

__device__ __forceinline__ v8f wmma_f16(v16h a, v16h b, v8f c) {
  // D = A(16x32 f16) * B(32x16 f16) + C(16x16 f32)
  return __builtin_amdgcn_wmma_f32_16x16x32_f16(
      /*neg_a=*/false, a, /*neg_b=*/false, b,
      /*c_mod=*/(short)0, c, /*reuse_a=*/false, /*reuse_b=*/false);
}

// A-frag (16x32, f16) from f32 memory where K is contiguous per row.
// Lane layout (ISA 7.12.2): lanes 0-15: elems 0..7 = K0..7, 8..15 = K16..23;
// lanes 16-31: elems 0..7 = K8..15, 8..15 = K24..31.
__device__ __forceinline__ v16h fragA_f32(const float* __restrict__ p, int lane) {
  const int off = (lane < 16) ? 0 : 8;
  float f0[8], f1[8];
#pragma unroll
  for (int i = 0; i < 8; ++i) f0[i] = p[off + i];
#pragma unroll
  for (int i = 0; i < 8; ++i) f1[i] = p[16 + off + i];
  v16h r;
#pragma unroll
  for (int i = 0; i < 8; ++i) r[i] = (_Float16)f0[i];
#pragma unroll
  for (int i = 0; i < 8; ++i) r[8 + i] = (_Float16)f1[i];
  return r;
}

// B-frag (32x16, f16) from f32 memory where K is contiguous for each column.
// lanes 0-15 hold K0..7,16..23, lanes 16-31 hold K8..15,24..31.
__device__ __forceinline__ v16h fragB_f32(const float* __restrict__ p, int lane) {
  const int off = (lane < 16) ? 0 : 8;
  float f[16];
#pragma unroll
  for (int i = 0; i < 16; ++i) f[i] = p[off + i + (i & 8)];
  v16h r;
#pragma unroll
  for (int i = 0; i < 16; ++i) r[i] = (_Float16)f[i];
  return r;
}

// ---------------------------------------------------------------------------
// f32 -> f16 weight conversion
// ---------------------------------------------------------------------------
__global__ __launch_bounds__(256) void f32_to_f16(const float* __restrict__ s,
                                                  _Float16* __restrict__ d, int n) {
  int i = blockIdx.x * 256 + threadIdx.x;
  if (i < n) d[i] = (_Float16)s[i];
}

__global__ __launch_bounds__(256) void zero_f32(float* __restrict__ p, long n) {
  long i = (long)blockIdx.x * 256 + threadIdx.x;
  if (i < n) p[i] = 0.0f;
}

// ---------------------------------------------------------------------------
// Generic WMMA GEMM:  O[z][m][n] (+)= sum_k A[m][k] * X_z[k][n] + bias[m]
//   A: [M][K] row-major f16 (preconverted weights)
//   X_z base = X + (z/xZDiv)*xZA + (z%xZDiv)*xZB ; row k stride = XROW (f32,
//   compile-time so staging loads become base+immediate and can clause)
//   O_z base = O + (z/oZDiv)*oZA + (z%oZDiv)*oZB ; row m stride = oRow (f32)
// Tiles: BM=64, BN=128, BK=32; 256 threads = 8 waves; each wave: 1 m-tile x 4 n-tiles.
// Requires M%64==0, N%128==0, K%32==0 (true for all calls here).
// ---------------------------------------------------------------------------
#define BM 64
#define BN 128
#define BK 32

template <int XROW>
__global__ __launch_bounds__(256) void wmma_gemm(
    const _Float16* __restrict__ A, const float* __restrict__ bias,
    const float* __restrict__ X, int xZDiv, long xZA, long xZB,
    float* __restrict__ O, int oZDiv, long oZA, long oZB, long oRow,
    int K, int accumulate) {
  // Row stride 40 halves = 80B (multiple of 16B so ds_load_b128 frags stay aligned)
  __shared__ __align__(16) _Float16 Xs[BN][BK + 8];

  const int tid  = threadIdx.x;
  const int lane = tid & 31;
  const int wave = tid >> 5;
  const int z    = blockIdx.z;

  const float* xb = X + (long)(z / xZDiv) * xZA + (long)(z % xZDiv) * xZB;
  float*       ob = O + (long)(z / oZDiv) * oZA + (long)(z % oZDiv) * oZB;

  const int n0blk = blockIdx.x * BN;
  const int m0    = blockIdx.y * BM + (wave >> 1) * 16;
  const int nw    = wave & 1;  // 0..1 -> 64-wide half of the n tile
  const int aoff  = (lane < 16) ? 0 : 8;
  const int boff  = (lane < 16) ? 0 : 8;
  const int lr    = lane & 15;

  // Staging geometry: this thread owns column nn and k-groups kg0/4 + u*8.
  const int nn  = tid & 127;
  const int kg0 = (tid >> 7) * 4;  // 0 or 4
  const float* xstage = xb + n0blk + nn + (long)kg0 * XROW;

  v8f acc[4];
#pragma unroll
  for (int j = 0; j < 4; ++j) acc[j] = v8f_zero();

  for (int k0 = 0; k0 < K; k0 += BK) {
    __syncthreads();

    // A fragment straight from global f16 weights (independent of LDS staging;
    // issued first so its latency overlaps the staging work below).
    FragU af;
    const _Float16* pa = A + (long)(m0 + lr) * K + k0 + aoff;
    af.u[0] = *(const uint4*)pa;
    af.u[1] = *(const uint4*)(pa + 16);

    // Stage 32(k) x 128(n) f32 tile -> LDS f16 [n][k-contiguous].
    // All 16 loads are base + compile-time immediate ((u*8+j)*XROW*4 fits the
    // signed 24-bit IOFFSET) -> single clause, one wait. Then 4 packed
    // ds_store_b64 per thread.
    const float* src = xstage + (long)k0 * XROW;
    float fst[4][4];
#pragma unroll
    for (int u = 0; u < 4; ++u)
#pragma unroll
      for (int j = 0; j < 4; ++j)
        fst[u][j] = src[(long)(u * 8 + j) * XROW];
#pragma unroll
    for (int u = 0; u < 4; ++u) {
      union { _Float16 h[4]; uint2 d; } pk;
#pragma unroll
      for (int j = 0; j < 4; ++j) pk.h[j] = (_Float16)fst[u][j];
      *(uint2*)&Xs[nn][kg0 + u * 8] = pk.d;  // 8B-aligned (80B rows)
    }

    // Prefetch next K tile (global_prefetch_b8)
    if (k0 + BK < K) {
      __builtin_prefetch(src + (long)BK * XROW, 0, 1);
    }
    __syncthreads();

    // Load all 4 B fragments, then issue the 4 WMMAs back-to-back.
    FragU bf[4];
#pragma unroll
    for (int j = 0; j < 4; ++j) {
      const int nloc = nw * 64 + j * 16 + lr;
      bf[j].u[0] = *(const uint4*)&Xs[nloc][boff];
      bf[j].u[1] = *(const uint4*)&Xs[nloc][boff + 16];
    }
#pragma unroll
    for (int j = 0; j < 4; ++j) acc[j] = wmma_f16(af.v, bf[j].v, acc[j]);
  }

  // Epilogue: C/D layout -> lanes 0-15: M=r, lanes 16-31: M=8+r; N=lane&15
#pragma unroll
  for (int j = 0; j < 4; ++j) {
    const int ng = n0blk + nw * 64 + j * 16 + lr;
#pragma unroll
    for (int r = 0; r < 8; ++r) {
      const int mg = m0 + ((lane < 16) ? r : 8 + r);
      float v = acc[j][r] + bias[mg];
      float* p = ob + (long)mg * oRow + ng;
      if (accumulate) v += *p;
      *p = v;
    }
  }
}

// ---------------------------------------------------------------------------
// Depthwise 3x3x3 conv, pad 1 (inter branch).  in/out: [NB][CH][T][H][W]
// ---------------------------------------------------------------------------
__global__ __launch_bounds__(256) void dwconv3d(
    const float* __restrict__ in, const float* __restrict__ wt,
    const float* __restrict__ bs, float* __restrict__ out,
    int NB, int CH, int T, int H, int W) {
  const long total = (long)NB * CH * T * H * W;
  long idx = (long)blockIdx.x * 256 + threadIdx.x;
  if (idx >= total) return;
  const int x = (int)(idx % W);  long t1 = idx / W;
  const int y = (int)(t1 % H);   long t2 = t1 / H;
  const int t = (int)(t2 % T);   long t3 = t2 / T;
  const int c = (int)(t3 % CH);
  const int z = (int)(t3 / CH);
  const float* ib = in + ((long)z * CH + c) * T * H * W;
  const float* wc = wt + c * 27;
  float acc = bs[c];
#pragma unroll
  for (int dt = -1; dt <= 1; ++dt) {
    const int tt = t + dt;
    if (tt < 0 || tt >= T) continue;
#pragma unroll
    for (int dy = -1; dy <= 1; ++dy) {
      const int yy = y + dy;
      if (yy < 0 || yy >= H) continue;
#pragma unroll
      for (int dx = -1; dx <= 1; ++dx) {
        const int xx = x + dx;
        if (xx < 0 || xx >= W) continue;
        acc += ib[((long)tt * H + yy) * W + xx] * wc[(dt + 1) * 9 + (dy + 1) * 3 + (dx + 1)];
      }
    }
  }
  out[idx] = acc;
}

// Depthwise 3x3 conv, pad 1 (intra branch).  in/out: [NB][CH][H][W]
__global__ __launch_bounds__(256) void dwconv2d(
    const float* __restrict__ in, const float* __restrict__ wt,
    const float* __restrict__ bs, float* __restrict__ out,
    int NB, int CH, int H, int W) {
  const long total = (long)NB * CH * H * W;
  long idx = (long)blockIdx.x * 256 + threadIdx.x;
  if (idx >= total) return;
  const int x = (int)(idx % W);  long t1 = idx / W;
  const int y = (int)(t1 % H);   long t2 = t1 / H;
  const int c = (int)(t2 % CH);
  const int z = (int)(t2 / CH);
  const float* ib = in + ((long)z * CH + c) * H * W;
  const float* wc = wt + c * 9;
  float acc = bs[c];
#pragma unroll
  for (int dy = -1; dy <= 1; ++dy) {
    const int yy = y + dy;
    if (yy < 0 || yy >= H) continue;
#pragma unroll
    for (int dx = -1; dx <= 1; ++dx) {
      const int xx = x + dx;
      if (xx < 0 || xx >= W) continue;
      acc += ib[(long)yy * W + xx] * wc[(dy + 1) * 3 + (dx + 1)];
    }
  }
  out[idx] = acc;
}

// ---------------------------------------------------------------------------
// 1 / max(||row||, eps) over the first 384 channels (q then k) of qkv.
// gridDim.x = NB*384; qkv: [NB][576][N]
// ---------------------------------------------------------------------------
__global__ __launch_bounds__(256) void row_invnorm(const float* __restrict__ qkv,
                                                   float* __restrict__ inv, int N) {
  __shared__ float red[256];
  const int r  = blockIdx.x;
  const int z  = r / 384;
  const int ch = r % 384;
  const float* p = qkv + ((long)z * 576 + ch) * (long)N;
  float ss = 0.0f;
  for (int i = threadIdx.x; i < N; i += 256) {
    const float v = p[i];
    ss += v * v;
  }
  red[threadIdx.x] = ss;
  __syncthreads();
  for (int s = 128; s > 0; s >>= 1) {
    if (threadIdx.x < s) red[threadIdx.x] += red[threadIdx.x + s];
    __syncthreads();
  }
  if (threadIdx.x == 0) inv[r] = 1.0f / fmaxf(sqrtf(red[0]), 1e-12f);
}

// ---------------------------------------------------------------------------
// Channel gram: partial S[z][h][32][32] += q_h(24xK) * k_h(24xK)^T via WMMA.
// q,k rows are K-contiguous -> direct-from-global fragments.
// grid: (K/KC, 8, NB), 1 wave per block; atomic f32 accumulate (K-split).
// Rows 24..31 read harmless in-buffer garbage; masked out in softmax.
// ---------------------------------------------------------------------------
__global__ __launch_bounds__(32) void gram_partial(const float* __restrict__ qkv,
                                                   float* __restrict__ S,
                                                   int N, int KC) {
  const int lane = threadIdx.x;
  const int z = blockIdx.z, h = blockIdx.y;
  const long zoff = (long)z * 576 * N;
  const float* qb = qkv + zoff;                  // q channels 0..191
  const float* kb = qkv + zoff + (long)192 * N;  // k channels
  const int kbase = blockIdx.x * KC;
  const int lr = lane & 15;

  const float* qrow0 = qb + (long)(h * 24 + lr) * N;
  const float* qrow1 = qb + (long)(h * 24 + 16 + lr) * N;
  const float* krow0 = kb + (long)(h * 24 + lr) * N;
  const float* krow1 = kb + (long)(h * 24 + 16 + lr) * N;

  v8f a00 = v8f_zero(), a01 = v8f_zero(), a10 = v8f_zero(), a11 = v8f_zero();

  for (int k0 = kbase; k0 < kbase + KC; k0 += 32) {
    const v16h fa0 = fragA_f32(qrow0 + k0, lane);
    const v16h fa1 = fragA_f32(qrow1 + k0, lane);
    const v16h fb0 = fragB_f32(krow0 + k0, lane);
    const v16h fb1 = fragB_f32(krow1 + k0, lane);
    a00 = wmma_f16(fa0, fb0, a00);
    a01 = wmma_f16(fa0, fb1, a01);
    a10 = wmma_f16(fa1, fb0, a10);
    a11 = wmma_f16(fa1, fb1, a11);
  }

  float* Sp = S + ((long)z * 8 + h) * 1024;
#pragma unroll
  for (int r = 0; r < 8; ++r) {
    const int mlo = (lane < 16) ? r : 8 + r;
    const int n0  = lane & 15;
    atomicAdd(&Sp[(mlo)      * 32 + n0],      a00[r]);
    atomicAdd(&Sp[(mlo)      * 32 + 16 + n0], a01[r]);
    atomicAdd(&Sp[(16 + mlo) * 32 + n0],      a10[r]);
    atomicAdd(&Sp[(16 + mlo) * 32 + 16 + n0], a11[r]);
  }
}

// ---------------------------------------------------------------------------
// Scale by invq*invk*temp, softmax over d (24), compact to attnW[zh][24][24].
// grid: NB*8 blocks of 32; lane = row c.
// ---------------------------------------------------------------------------
__global__ __launch_bounds__(32) void softmax_attn(const float* __restrict__ S,
                                                   const float* __restrict__ inv,
                                                   const float* __restrict__ temp,
                                                   float* __restrict__ attnW) {
  const int bid = blockIdx.x;
  const int z = bid / 8, h = bid % 8;
  const int c = threadIdx.x;
  if (c >= 24) return;
  const float* Sp   = S + (long)bid * 1024;
  const float* invz = inv + z * 384;
  const float tv = temp[h];
  const float iq = invz[h * 24 + c];
  float row[24];
  float mx = -3.4e38f;
#pragma unroll
  for (int d = 0; d < 24; ++d) {
    const float v = Sp[c * 32 + d] * iq * invz[192 + h * 24 + d] * tv;
    row[d] = v;
    mx = fmaxf(mx, v);
  }
  float sum = 0.0f;
#pragma unroll
  for (int d = 0; d < 24; ++d) {
    const float e = __expf(row[d] - mx);
    row[d] = e;
    sum += e;
  }
  const float rs = 1.0f / sum;
  float* outp = attnW + (long)bid * 576;
#pragma unroll
  for (int d = 0; d < 24; ++d) outp[c * 24 + d] = row[d] * rs;
}

// ---------------------------------------------------------------------------
// o[z][h*24+cc][n] = sum_d attn[z][h][cc][d] * v[z][384+h*24+d][n]
// grid: (N/256, 8, NB), 256 threads; v tile staged in LDS (reused 24x).
// ---------------------------------------------------------------------------
__global__ __launch_bounds__(256) void apply_attn(const float* __restrict__ qkv,
                                                  const float* __restrict__ attnW,
                                                  float* __restrict__ o, int N) {
  __shared__ float vt[24][257];
  __shared__ float aw[576];
  const int z = blockIdx.z, h = blockIdx.y;
  const int n0 = blockIdx.x * 256;
  const int t = threadIdx.x;
  const float* vb = qkv + ((long)z * 576 + 384 + h * 24) * (long)N;
#pragma unroll
  for (int d = 0; d < 24; ++d) vt[d][t] = vb[(long)d * N + n0 + t];
  for (int i = t; i < 576; i += 256) aw[i] = attnW[((long)z * 8 + h) * 576 + i];
  __syncthreads();
  float* ob = o + ((long)z * 192 + h * 24) * (long)N + n0 + t;
#pragma unroll 4
  for (int cc = 0; cc < 24; ++cc) {
    float acc = 0.0f;
#pragma unroll
    for (int d = 0; d < 24; ++d) acc += aw[cc * 24 + d] * vt[d][t];
    ob[(long)cc * N] = acc;
  }
}

// ---------------------------------------------------------------------------
// Host-side orchestration
// ---------------------------------------------------------------------------
extern "C" void kernel_launch(void* const* d_in, const int* in_sizes, int n_in,
                              void* d_out, int out_size, void* d_ws, size_t ws_size,
                              hipStream_t stream) {
  (void)in_sizes; (void)n_in; (void)out_size; (void)ws_size;

  const float* x      = (const float*)d_in[0];
  const float* temp   = (const float*)d_in[1];
  const float* wQkvI  = (const float*)d_in[2];
  const float* bQkvI  = (const float*)d_in[3];
  const float* wDwI   = (const float*)d_in[4];
  const float* bDwI   = (const float*)d_in[5];
  const float* wProjI = (const float*)d_in[6];
  const float* bProjI = (const float*)d_in[7];
  const float* wQkvA  = (const float*)d_in[8];
  const float* bQkvA  = (const float*)d_in[9];
  const float* wDwA   = (const float*)d_in[10];
  const float* bDwA   = (const float*)d_in[11];
  const float* wProjA = (const float*)d_in[12];
  const float* bProjA = (const float*)d_in[13];
  float* out = (float*)d_out;

  constexpr int  B = 2, C = 192, T = 8, H = 64, W = 64, HEADS = 8;
  constexpr long N3 = (long)T * H * W;  // 32768
  constexpr long N2 = (long)H * W;      // 4096
  constexpr int  CH = 3 * C;            // 576

  // Workspace carve-up (256B aligned slices)
  char* ws = (char*)d_ws;
  size_t off = 0;
  auto alloc = [&](size_t bytes) -> char* {
    char* p = ws + off;
    off = (off + bytes + 255) & ~(size_t)255;
    return p;
  };
  _Float16* hQkvI  = (_Float16*)alloc((size_t)CH * C * sizeof(_Float16));
  _Float16* hProjI = (_Float16*)alloc((size_t)C * C * sizeof(_Float16));
  _Float16* hQkvA  = (_Float16*)alloc((size_t)CH * C * sizeof(_Float16));
  _Float16* hProjA = (_Float16*)alloc((size_t)C * C * sizeof(_Float16));
  float* invI  = (float*)alloc((size_t)B * 384 * 4);
  float* invA  = (float*)alloc((size_t)B * T * 384 * 4);
  float* Sb    = (float*)alloc((size_t)B * T * HEADS * 1024 * 4);
  float* attnB = (float*)alloc((size_t)B * T * HEADS * 576 * 4);
  float* buf1  = (float*)alloc((size_t)B * CH * N3 * 4);  // qkv pre-DW
  float* buf2  = (float*)alloc((size_t)B * CH * N3 * 4);  // qkv post-DW
  float* buf3  = (float*)alloc((size_t)B * C * N3 * 4);   // attention output

  // 0. Convert weights to f16
  f32_to_f16<<<(CH * C + 255) / 256, 256, 0, stream>>>(wQkvI, hQkvI, CH * C);
  f32_to_f16<<<(C * C + 255) / 256, 256, 0, stream>>>(wProjI, hProjI, C * C);
  f32_to_f16<<<(CH * C + 255) / 256, 256, 0, stream>>>(wQkvA, hQkvA, CH * C);
  f32_to_f16<<<(C * C + 255) / 256, 256, 0, stream>>>(wProjA, hProjA, C * C);

  // ===== inter (3D) branch =====
  // 1. qkv 1x1x1: [B][576][N3] = W(576x192) @ x_b(192xN3)
  wmma_gemm<32768><<<dim3((unsigned)(N3 / BN), CH / BM, B), 256, 0, stream>>>(
      hQkvI, bQkvI, x, 1, (long)C * N3, 0,
      buf1, 1, (long)CH * N3, 0, N3, C, 0);
  // 2. depthwise 3x3x3
  {
    const long total = (long)B * CH * N3;
    dwconv3d<<<(unsigned)((total + 255) / 256), 256, 0, stream>>>(
        buf1, wDwI, bDwI, buf2, B, CH, T, H, W);
  }
  // 3. row inverse norms for q,k
  row_invnorm<<<B * 384, 256, 0, stream>>>(buf2, invI, (int)N3);
  // 4. gram + softmax
  zero_f32<<<(unsigned)((B * HEADS * 1024 + 255) / 256), 256, 0, stream>>>(
      Sb, (long)B * HEADS * 1024);
  gram_partial<<<dim3((unsigned)(N3 / 512), HEADS, B), 32, 0, stream>>>(
      buf2, Sb, (int)N3, 512);
  softmax_attn<<<B * HEADS, 32, 0, stream>>>(Sb, invI, temp, attnB);
  // 5. o = attn @ v
  apply_attn<<<dim3((unsigned)(N3 / 256), HEADS, B), 256, 0, stream>>>(
      buf2, attnB, buf3, (int)N3);
  // 6. proj 1x1x1 -> d_out (overwrite)
  wmma_gemm<32768><<<dim3((unsigned)(N3 / BN), C / BM, B), 256, 0, stream>>>(
      hProjI, bProjI, buf3, 1, (long)C * N3, 0,
      out, 1, (long)C * N3, 0, N3, C, 0);

  // ===== intra (per-frame 2D) branch =====
  const int NF = B * T;  // 16 frames
  // 7. qkv 1x1 per frame: frame z -> x base (z/T)*C*N3 + (z%T)*N2, row stride N3
  wmma_gemm<32768><<<dim3((unsigned)(N2 / BN), CH / BM, NF), 256, 0, stream>>>(
      hQkvA, bQkvA, x, T, (long)C * N3, N2,
      buf1, 1, (long)CH * N2, 0, N2, C, 0);
  // 8. depthwise 3x3
  {
    const long total = (long)NF * CH * N2;
    dwconv2d<<<(unsigned)((total + 255) / 256), 256, 0, stream>>>(
        buf1, wDwA, bDwA, buf2, NF, CH, H, W);
  }
  // 9. norms, gram, softmax
  row_invnorm<<<NF * 384, 256, 0, stream>>>(buf2, invA, (int)N2);
  zero_f32<<<(unsigned)((NF * HEADS * 1024 + 255) / 256), 256, 0, stream>>>(
      Sb, (long)NF * HEADS * 1024);
  gram_partial<<<dim3((unsigned)(N2 / 512), HEADS, NF), 32, 0, stream>>>(
      buf2, Sb, (int)N2, 512);
  softmax_attn<<<NF * HEADS, 32, 0, stream>>>(Sb, invA, temp, attnB);
  // 10. o2 = attn @ v
  apply_attn<<<dim3((unsigned)(N2 / 256), HEADS, NF), 256, 0, stream>>>(
      buf2, attnB, buf3, (int)N2);
  // 11. proj 1x1 with transposed accumulate into d_out:
  //     frame z=(b*T+t): out offset = b*C*N3 + t*N2 + oc*N3 + n
  wmma_gemm<4096><<<dim3((unsigned)(N2 / BN), C / BM, NF), 256, 0, stream>>>(
      hProjA, bProjA, buf3, 1, (long)C * N2, 0,
      out, T, (long)C * N3, N2, N3, C, 1);
}
